// VGG_Trans_88716844466761
// MI455X (gfx1250) — matrix operand order repaired and checked by reading, not compiled, not run on hardware
//
#include <hip/hip_runtime.h>
#include <hip/hip_bf16.h>

// ---------------------------------------------------------------------------
// MI455X (gfx1250, wave32). All heavy math on v_wmma_f32_16x16x32_bf16.
// A operands: row-major bf16, fetched as 2x b128 per lane (lane half h holds
// the 16-element K run 16h..16h+15, same order as B). B operands: pre-swizzled into
// per-lane fragment order (one 32B contiguous load). Each wave computes a
// 16x64 output tile: 4 WMMAs per A fragment.
// Workspace assumption: ws_size >= ~110 MB.
// ---------------------------------------------------------------------------

typedef __attribute__((ext_vector_type(16))) __bf16 v16bf;
typedef __attribute__((ext_vector_type(8)))  __bf16 v8bf;
typedef __attribute__((ext_vector_type(8)))  float  v8f;

static __device__ __forceinline__ float bf2f(__bf16 b) {
  unsigned short s = __builtin_bit_cast(unsigned short, b);
  unsigned int u = ((unsigned int)s) << 16;
  return __builtin_bit_cast(float, u);
}
static __device__ __forceinline__ __bf16 f2bf(float f) {
  unsigned int u = __builtin_bit_cast(unsigned int, f);
  u += 0x7fffu + ((u >> 16) & 1u);  // round-to-nearest-even
  unsigned short s = (unsigned short)(u >> 16);
  return __builtin_bit_cast(__bf16, s);
}
static __device__ __forceinline__ __bf16 bf_zero() {
  return __builtin_bit_cast(__bf16, (unsigned short)0);
}
static __device__ __forceinline__ v16bf cat8(v8bf lo, v8bf hi) {
  return __builtin_shufflevector(lo, hi, 0, 1, 2, 3, 4, 5, 6, 7,
                                 8, 9, 10, 11, 12, 13, 14, 15);
}

// ---------------------------------------------------------------------------
// WMMA GEMM:  C[M,N] = relu?( scale * (A[M,K] x B[K,N]) + bias )
// A: row-major bf16 (lda), 16B-aligned rows, K % 32 == 0.
// B: swizzled fragment order: [kp][nt][lane][16], kp = K/32, nt = N/16.
// One wave -> 16x64 tile (N % 64 == 0). Optional f32 / bf16 outputs,
// optional attention key-mask epilogue.
// ---------------------------------------------------------------------------
__global__ void __launch_bounds__(256) k_gemm(
    const __bf16* __restrict__ A, int lda,
    const __bf16* __restrict__ Bsw,
    const float* __restrict__ bias,
    float* __restrict__ C, int ldc,
    __bf16* __restrict__ Cb, int ldcb,
    int M, int K, int N, float scale, int relu,
    const float* __restrict__ keymask) {
  int nt4 = N >> 6;
  int wave = blockIdx.x * 8 + (threadIdx.x >> 5);
  int total = (M >> 4) * nt4;
  if (wave >= total) return;
  int m0 = (wave / nt4) << 4;
  int n0 = (wave % nt4) << 6;
  int lane = threadIdx.x & 31;
  int hh = lane >> 4, l15 = lane & 15;
  int ntiles = N >> 4;
  v8f acc0 = {}, acc1 = {}, acc2 = {}, acc3 = {};
  const __bf16* Ap = A + (size_t)(m0 + l15) * lda + hh * 16;
  const __bf16* Bp = Bsw + ((size_t)(n0 >> 4) * 32 + lane) * 16;
  size_t bpanel = (size_t)ntiles * 512;  // elems per K-panel
  for (int k0 = 0; k0 < K; k0 += 32) {
    v8bf lo = *(const v8bf*)(Ap + k0);
    v8bf hi = *(const v8bf*)(Ap + k0 + 8);
    v16bf a = cat8(lo, hi);
    const __bf16* Bk = Bp + (size_t)(k0 >> 5) * bpanel;
    __builtin_prefetch(Bk + bpanel, 0, 1);
    v16bf b0 = *(const v16bf*)(Bk);
    v16bf b1 = *(const v16bf*)(Bk + 512);
    v16bf b2 = *(const v16bf*)(Bk + 1024);
    v16bf b3 = *(const v16bf*)(Bk + 1536);
    acc0 = __builtin_amdgcn_wmma_f32_16x16x32_bf16(false, a, false, b0, (short)0, acc0, false, false);
    acc1 = __builtin_amdgcn_wmma_f32_16x16x32_bf16(false, a, false, b1, (short)0, acc1, false, false);
    acc2 = __builtin_amdgcn_wmma_f32_16x16x32_bf16(false, a, false, b2, (short)0, acc2, false, false);
    acc3 = __builtin_amdgcn_wmma_f32_16x16x32_bf16(false, a, false, b3, (short)0, acc3, false, false);
  }
  v8f accs[4] = {acc0, acc1, acc2, acc3};
#pragma unroll
  for (int j = 0; j < 4; ++j) {
    int n = n0 + j * 16 + l15;
#pragma unroll
    for (int r = 0; r < 8; ++r) {
      int m = m0 + r + (hh << 3);
      float v = accs[j][r] * scale;
      if (bias) v += bias[n];
      if (keymask && keymask[n] == 0.0f) v = -1e9f;
      if (relu) v = fmaxf(v, 0.0f);
      if (C)  C[(size_t)m * ldc + n] = v;
      if (Cb) Cb[(size_t)m * ldcb + n] = f2bf(v);
    }
  }
}

// ---------------------------------------------------------------------------
// Implicit-GEMM 3x3 conv, pad=1.  Activations HWC bf16; weights swizzled
// fragment order over K = 9*Cin (zero-padded to mult of 32), N = Cout.
// Vector A path (b128 per 8-wide K chunk) when Cin % 8 == 0 (then
// 9*Cin % 32 == 0 too for Cin in {64,128,256,512}); scalar gather otherwise
// (the 3-channel stem).  One wave -> 16x64 tile (Cout % 64 == 0).
// ---------------------------------------------------------------------------
__global__ void __launch_bounds__(256) k_conv3x3(
    const __bf16* __restrict__ act, const __bf16* __restrict__ Bsw,
    const float* __restrict__ bias, __bf16* __restrict__ out,
    int H, int W, int Cin, int Cout, int relu) {
  int nt4 = Cout >> 6;
  int wave = blockIdx.x * 8 + (threadIdx.x >> 5);
  int total = ((H * W) >> 4) * nt4;
  if (wave >= total) return;
  int m0 = (wave / nt4) << 4;
  int n0 = (wave % nt4) << 6;
  int lane = threadIdx.x & 31;
  int hh = lane >> 4, l15 = lane & 15;
  int m = m0 + l15;
  int oy = m / W, ox = m - oy * W;
  int Ktot = 9 * Cin;
  int Kpad = (Ktot + 31) & ~31;
  int ntiles = Cout >> 4;
  int vecA = ((Cin & 7) == 0);
  v8f acc0 = {}, acc1 = {}, acc2 = {}, acc3 = {};
  const __bf16* Bp = Bsw + ((size_t)(n0 >> 4) * 32 + lane) * 16;
  size_t bpanel = (size_t)ntiles * 512;
  for (int k0 = 0; k0 < Kpad; k0 += 32) {
    v16bf a;
    if (vecA) {
      v8bf ch[2];
#pragma unroll
      for (int c = 0; c < 2; ++c) {
        int kb = k0 + hh * 16 + c * 8;
        v8bf v = {};
        int kp = kb / Cin, ci = kb - kp * Cin;
        int ky = kp / 3, kx = kp - ky * 3;
        int iy = oy + ky - 1, ix = ox + kx - 1;
        if (iy >= 0 && iy < H && ix >= 0 && ix < W)
          v = *(const v8bf*)(act + ((size_t)iy * W + ix) * Cin + ci);
        ch[c] = v;
      }
      a = cat8(ch[0], ch[1]);
    } else {
#pragma unroll
      for (int i = 0; i < 16; ++i) {
        int vg = i >> 1, e = i & 1;
        int kl = (hh << 4) + (vg << 1) + e;
        int kg = k0 + kl;
        __bf16 v = bf_zero();
        if (kg < Ktot) {
          int kp = kg / Cin, ci = kg - kp * Cin;
          int ky = kp / 3, kx = kp - ky * 3;
          int iy = oy + ky - 1, ix = ox + kx - 1;
          if (iy >= 0 && iy < H && ix >= 0 && ix < W)
            v = act[((size_t)iy * W + ix) * Cin + ci];
        }
        a[i] = v;
      }
    }
    const __bf16* Bk = Bp + (size_t)(k0 >> 5) * bpanel;
    __builtin_prefetch(Bk + bpanel, 0, 1);
    v16bf b0 = *(const v16bf*)(Bk);
    v16bf b1 = *(const v16bf*)(Bk + 512);
    v16bf b2 = *(const v16bf*)(Bk + 1024);
    v16bf b3 = *(const v16bf*)(Bk + 1536);
    acc0 = __builtin_amdgcn_wmma_f32_16x16x32_bf16(false, a, false, b0, (short)0, acc0, false, false);
    acc1 = __builtin_amdgcn_wmma_f32_16x16x32_bf16(false, a, false, b1, (short)0, acc1, false, false);
    acc2 = __builtin_amdgcn_wmma_f32_16x16x32_bf16(false, a, false, b2, (short)0, acc2, false, false);
    acc3 = __builtin_amdgcn_wmma_f32_16x16x32_bf16(false, a, false, b3, (short)0, acc3, false, false);
  }
  v8f accs[4] = {acc0, acc1, acc2, acc3};
#pragma unroll
  for (int j = 0; j < 4; ++j) {
    int n = n0 + j * 16 + l15;
#pragma unroll
    for (int r = 0; r < 8; ++r) {
      int mm = m0 + r + (hh << 3);
      float v = accs[j][r] + bias[n];
      if (relu) v = fmaxf(v, 0.0f);
      out[(size_t)mm * Cout + n] = f2bf(v);
    }
  }
}

// ---------------------------------------------------------------------------
// Repack kernels: produce B operands in swizzled fragment order
//   dst[((kp*ntiles + nt)*32 + lane)*16 + i] = B[kp*32 + (lane>>4)*16 + i]
//                                               [nt*16 + (lane&15)]
// ---------------------------------------------------------------------------
// linear weight [Out][In] f32 -> swizzled B (K=In, N=Out)
__global__ void k_wlin(const float* __restrict__ w, __bf16* __restrict__ dst,
                       int Out, int In) {
  size_t idx = (size_t)blockIdx.x * blockDim.x + threadIdx.x;
  if (idx >= (size_t)Out * In) return;
  int ntiles = Out >> 4;
  size_t pano = idx >> 9;
  int li = (idx >> 4) & 31, i = idx & 15;
  int kp = pano / ntiles, nt = pano - (size_t)kp * ntiles;
  int K = kp * 32 + ((li >> 4) << 4) + i;
  int n = nt * 16 + (li & 15);
  dst[idx] = f2bf(w[(size_t)n * In + K]);
}

// conv weight [Cout][Cin][3][3] f32 -> swizzled B (K = 9*Cin padded, N = Cout)
__global__ void k_wconv(const float* __restrict__ w, __bf16* __restrict__ dst,
                        int Cout, int Cin) {
  int Ktot = 9 * Cin;
  int Kpad = (Ktot + 31) & ~31;
  size_t n_elem = (size_t)Kpad * Cout;
  size_t idx = (size_t)blockIdx.x * blockDim.x + threadIdx.x;
  if (idx >= n_elem) return;
  int ntiles = Cout >> 4;
  size_t pano = idx >> 9;
  int li = (idx >> 4) & 31, i = idx & 15;
  int kp = pano / ntiles, nt = pano - (size_t)kp * ntiles;
  int K = kp * 32 + ((li >> 4) << 4) + i;
  int n = nt * 16 + (li & 15);
  __bf16 v = bf_zero();
  if (K < Ktot) {
    int kpos = K / Cin, ci = K - kpos * Cin;
    v = f2bf(w[((size_t)n * Cin + ci) * 9 + kpos]);
  }
  dst[idx] = v;
}

// generic f32 row-major [K][ld] (+coloff) -> swizzled B (KxN)
__global__ void k_packB(const float* __restrict__ src, int ld, int coloff,
                        int K, int N, __bf16* __restrict__ dst) {
  size_t idx = (size_t)blockIdx.x * blockDim.x + threadIdx.x;
  if (idx >= (size_t)K * N) return;
  int ntiles = N >> 4;
  size_t pano = idx >> 9;
  int li = (idx >> 4) & 31, i = idx & 15;
  int kp = pano / ntiles, nt = pano - (size_t)kp * ntiles;
  int kk = kp * 32 + ((li >> 4) << 4) + i;
  int n = nt * 16 + (li & 15);
  dst[idx] = f2bf(src[(size_t)kk * ld + coloff + n]);
}

// attention K^T: kf [T=1024][512] f32 -> per-head swizzled B (K=256, N=1024)
__global__ void k_packKT(const float* __restrict__ kf,
                         __bf16* __restrict__ dst) {
  size_t idx = (size_t)blockIdx.x * blockDim.x + threadIdx.x;
  if (idx >= 2ull * 256 * 1024) return;
  int h = idx / (256 * 1024);
  size_t r = idx - (size_t)h * 256 * 1024;
  size_t pano = r >> 9;
  int li = (r >> 4) & 31, i = r & 15;
  int kp = pano >> 6, nt = pano & 63;  // ntiles = 1024/16 = 64
  int K = kp * 32 + ((li >> 4) << 4) + i;
  int n = nt * 16 + (li & 15);
  dst[idx] = f2bf(kf[(size_t)n * 512 + h * 256 + K]);
}

// ---------------------------------------------------------------------------
// Elementwise / small kernels
// ---------------------------------------------------------------------------
__global__ void k_in_hwc(const float* __restrict__ x, __bf16* __restrict__ out,
                         int C, int HW) {
  size_t idx = (size_t)blockIdx.x * blockDim.x + threadIdx.x;
  if (idx >= (size_t)C * HW) return;
  int c = idx % C;
  size_t p = idx / C;
  out[idx] = f2bf(x[(size_t)c * HW + p]);
}

__global__ void k_maxpool(const __bf16* __restrict__ in,
                          __bf16* __restrict__ out, int H, int W, int C) {
  size_t idx = (size_t)blockIdx.x * blockDim.x + threadIdx.x;
  int Ho = H >> 1, Wo = W >> 1;
  if (idx >= (size_t)Ho * Wo * C) return;
  int c = idx % C;
  size_t p = idx / C;
  int xo = p % Wo, yo = p / Wo;
  const __bf16* r0 = in + (((size_t)(2 * yo) * W) + 2 * xo) * C + c;
  const __bf16* r1 = r0 + (size_t)W * C;
  float v = fmaxf(fmaxf(bf2f(r0[0]), bf2f(r0[C])),
                  fmaxf(bf2f(r1[0]), bf2f(r1[C])));
  out[idx] = f2bf(v);
}

__global__ void k_bf2f(const __bf16* __restrict__ in, float* __restrict__ out,
                       size_t n) {
  size_t i = (size_t)blockIdx.x * blockDim.x + threadIdx.x;
  if (i < n) out[i] = bf2f(in[i]);
}
__global__ void k_f2bf(const float* __restrict__ in, __bf16* __restrict__ out,
                       size_t n) {
  size_t i = (size_t)blockIdx.x * blockDim.x + threadIdx.x;
  if (i < n) out[i] = f2bf(in[i]);
}

__global__ void k_select(float* __restrict__ y, const float* __restrict__ tok,
                         const float* __restrict__ selF, int C, size_t n) {
  size_t i = (size_t)blockIdx.x * blockDim.x + threadIdx.x;
  if (i >= n) return;
  size_t t = i / C;
  if (selF[t] == 0.0f) y[i] = tok[i];
}

// 1x1 conv with Cout=1; mode 1: sigmoid -> out & selF; mode 2: relu*selF
__global__ void __launch_bounds__(128) k_conv1x1_c1(
    const __bf16* __restrict__ act, const float* __restrict__ w,
    const float* __restrict__ b, float* __restrict__ out, int Cin, int mode,
    float* __restrict__ selF) {
  int p = blockIdx.x;
  int t = threadIdx.x;
  float s = 0.0f;
  for (int c = t; c < Cin; c += 128) s += bf2f(act[(size_t)p * Cin + c]) * w[c];
  __shared__ float red[128];
  red[t] = s;
  __syncthreads();
  for (int o = 64; o > 0; o >>= 1) {
    if (t < o) red[t] += red[t + o];
    __syncthreads();
  }
  if (t == 0) {
    float v = red[0] + b[0];
    if (mode == 1) {
      float sg = 1.0f / (1.0f + __expf(-v));
      out[p] = sg;
      selF[p] = sg > 0.5f ? 1.0f : 0.0f;
    } else if (mode == 2) {
      out[p] = fmaxf(v, 0.0f) * selF[p];
    } else {
      out[p] = v;
    }
  }
}

__global__ void __launch_bounds__(256) k_softmax(const float* __restrict__ S,
                                                 __bf16* __restrict__ P,
                                                 int N) {
  int row = blockIdx.x, t = threadIdx.x;
  const float* s = S + (size_t)row * N;
  __bf16* p = P + (size_t)row * N;
  __shared__ float red[256];
  float mx = -3.4e38f;
  for (int i = t; i < N; i += 256) mx = fmaxf(mx, s[i]);
  red[t] = mx;
  __syncthreads();
  for (int o = 128; o > 0; o >>= 1) {
    if (t < o) red[t] = fmaxf(red[t], red[t + o]);
    __syncthreads();
  }
  mx = red[0];
  __syncthreads();
  float sum = 0.0f;
  for (int i = t; i < N; i += 256) sum += __expf(s[i] - mx);
  red[t] = sum;
  __syncthreads();
  for (int o = 128; o > 0; o >>= 1) {
    if (t < o) red[t] += red[t + o];
    __syncthreads();
  }
  float inv = 1.0f / red[0];
  for (int i = t; i < N; i += 256) p[i] = f2bf(__expf(s[i] - mx) * inv);
}

__global__ void __launch_bounds__(256) k_add_ln(float* __restrict__ y,
                                                const float* __restrict__ t2,
                                                const float* __restrict__ g,
                                                const float* __restrict__ b) {
  const int C = 512;
  int row = blockIdx.x, t = threadIdx.x;
  float* yr = y + (size_t)row * C;
  const float* tr = t2 + (size_t)row * C;
  __shared__ float red[256];
  float v0 = yr[t] + tr[t];
  float v1 = yr[t + 256] + tr[t + 256];
  red[t] = v0 + v1;
  __syncthreads();
  for (int o = 128; o > 0; o >>= 1) {
    if (t < o) red[t] += red[t + o];
    __syncthreads();
  }
  float mean = red[0] / C;
  __syncthreads();
  float d0 = v0 - mean, d1 = v1 - mean;
  red[t] = d0 * d0 + d1 * d1;
  __syncthreads();
  for (int o = 128; o > 0; o >>= 1) {
    if (t < o) red[t] += red[t + o];
    __syncthreads();
  }
  float inv = rsqrtf(red[0] / C + 1e-5f);
  yr[t] = d0 * inv * g[t] + b[t];
  yr[t + 256] = d1 * inv * g[t + 256] + b[t + 256];
}

// ---------------------------------------------------------------------------
// Host orchestration
// ---------------------------------------------------------------------------
static inline int ceil_div(long long a, long long b) { return (int)((a + b - 1) / b); }

extern "C" void kernel_launch(void* const* d_in, const int* in_sizes, int n_in,
                              void* d_out, int out_size, void* d_ws,
                              size_t ws_size, hipStream_t stream) {
  (void)in_sizes; (void)n_in; (void)out_size; (void)ws_size;
  const size_t MB = 1024ull * 1024ull;
  char* ws = (char*)d_ws;
  __bf16* actA   = (__bf16*)(ws + 0 * MB);    // 34 MB max
  __bf16* actB   = (__bf16*)(ws + 36 * MB);   // 34 MB max
  __bf16* wbuf   = (__bf16*)(ws + 72 * MB);   // 4.8 MB max (swizzled weights)
  float*  scores = (float*) (ws + 78 * MB);   // 8 MB
  __bf16* probs  = (__bf16*)(ws + 86 * MB);   // 4 MB
  float*  tokf   = (float*) (ws + 90 * MB);   // 2 MB
  float*  yf     = (float*) (ws + 92 * MB);   // 2 MB
  float*  t2f    = (float*) (ws + 94 * MB);   // 2 MB
  float*  kf     = (float*) (ws + 96 * MB);   // 2 MB
  __bf16* qbf    = (__bf16*)(ws + 98 * MB);   // 1 MB  (row-major A)
  __bf16* kTsw   = (__bf16*)(ws + 99 * MB);   // 1 MB  (swizzled B, 2 heads)
  __bf16* vsw    = (__bf16*)(ws + 100 * MB);  // 1 MB  (swizzled B, 2 heads)
  __bf16* ybf    = (__bf16*)(ws + 101 * MB);  // 1 MB
  __bf16* obf    = (__bf16*)(ws + 102 * MB);  // 1 MB
  __bf16* ffnbf  = (__bf16*)(ws + 103 * MB);  // 2 MB
  float*  selF   = (float*) (ws + 105 * MB);  // 4 KB
  __bf16* h1     = (__bf16*)(ws + 106 * MB);  // 0.5 MB
  __bf16* h2     = (__bf16*)(ws + 107 * MB);  // 0.25 MB
  float*  vf     = (float*) (ws + 108 * MB);  // 2 MB

  const float* x = (const float*)d_in[0];
  auto F = [&](int i) { return (const float*)d_in[i]; };
  const int MASK0 = 27, REG0 = 33, DEC0 = 39;
  const int T = 1024, C = 512;

  auto gemm = [&](const __bf16* A, int lda, const __bf16* Bsw,
                  const float* bias, float* Co, int ldc, __bf16* Cb, int ldcb,
                  int M, int K, int N, float scale, int relu,
                  const float* mask) {
    int waves = (M / 16) * (N / 64);
    k_gemm<<<ceil_div(waves, 8), 256, 0, stream>>>(A, lda, Bsw, bias, Co, ldc,
                                                   Cb, ldcb, M, K, N, scale,
                                                   relu, mask);
  };
  auto conv3 = [&](const __bf16* act, const float* w, const float* b,
                   __bf16* out, int H, int W, int Ci, int Co, int relu) {
    int Kpad = (9 * Ci + 31) & ~31;
    long long nw = (long long)Kpad * Co;
    k_wconv<<<ceil_div(nw, 256), 256, 0, stream>>>(w, wbuf, Co, Ci);
    int waves = (H * W / 16) * (Co / 64);
    k_conv3x3<<<ceil_div(waves, 8), 256, 0, stream>>>(act, wbuf, b, out, H, W,
                                                      Ci, Co, relu);
  };

  // ---------------- VGG backbone ----------------
  int H = 512, W = 512, Cin = 3, wi = 0;
  __bf16 *cur = actA, *nxt = actB;
  k_in_hwc<<<ceil_div(3LL * 512 * 512, 256), 256, 0, stream>>>(x, cur, 3,
                                                               512 * 512);
  static const int CFGI[17] = {64, 64, 0, 128, 128, 0, 256, 256, 256, 0,
                               512, 512, 512, 0, 512, 512, 512};
  for (int s = 0; s < 17; ++s) {
    if (CFGI[s] == 0) {
      long long n = (long long)(H / 2) * (W / 2) * Cin;
      k_maxpool<<<ceil_div(n, 256), 256, 0, stream>>>(cur, nxt, H, W, Cin);
      H /= 2; W /= 2;
    } else {
      int Co = CFGI[s];
      conv3(cur, F(1 + 2 * wi), F(2 + 2 * wi), nxt, H, W, Cin, Co, 1);
      Cin = Co; ++wi;
    }
    __bf16* t = cur; cur = nxt; nxt = t;
  }
  __bf16* tokbf = cur;  // final feature f, bf16 HWC [1024][512]

  // ---------------- mask head -> mask, selF ----------------
  conv3(tokbf, F(MASK0 + 0), F(MASK0 + 1), h1, 32, 32, 512, 256, 1);
  conv3(h1, F(MASK0 + 2), F(MASK0 + 3), h2, 32, 32, 256, 128, 1);
  k_conv1x1_c1<<<1024, 128, 0, stream>>>(h2, F(MASK0 + 4), F(MASK0 + 5),
                                         (float*)d_out + 1024, 128, 1, selF);

  // ---------------- tok / y init ----------------
  size_t TC = (size_t)T * C;
  k_bf2f<<<ceil_div(TC, 256), 256, 0, stream>>>(tokbf, tokf, TC);
  k_bf2f<<<ceil_div(TC, 256), 256, 0, stream>>>(tokbf, yf, TC);
  k_bf2f<<<ceil_div(TC, 256), 256, 0, stream>>>(tokbf, (float*)d_out + 2048, TC);

  // ---------------- decoder layers ----------------
  for (int L = 0; L < 2; ++L) {
    int B0 = DEC0 + 26 * L;
    auto mha = [&](const __bf16* qsrc, const __bf16* kvsrc, int pbase,
                   const float* kmask) {
      long long nl = 512LL * 512;
      // q -> qbf (row-major A for scores)
      k_wlin<<<ceil_div(nl, 256), 256, 0, stream>>>(F(pbase + 0), wbuf, 512, 512);
      gemm(qsrc, 512, wbuf, F(pbase + 1), nullptr, 0, qbf, 512,
           T, 512, 512, 1.0f, 0, nullptr);
      // k -> kf f32 -> per-head swizzled K^T
      k_wlin<<<ceil_div(nl, 256), 256, 0, stream>>>(F(pbase + 2), wbuf, 512, 512);
      gemm(kvsrc, 512, wbuf, F(pbase + 3), kf, 512, nullptr, 0,
           T, 512, 512, 1.0f, 0, nullptr);
      k_packKT<<<ceil_div(2LL * 256 * 1024, 256), 256, 0, stream>>>(kf, kTsw);
      // v -> vf f32 -> per-head swizzled B
      k_wlin<<<ceil_div(nl, 256), 256, 0, stream>>>(F(pbase + 4), wbuf, 512, 512);
      gemm(kvsrc, 512, wbuf, F(pbase + 5), vf, 512, nullptr, 0,
           T, 512, 512, 1.0f, 0, nullptr);
      for (int h = 0; h < 2; ++h)
        k_packB<<<ceil_div(1024LL * 256, 256), 256, 0, stream>>>(
            vf, 512, h * 256, 1024, 256, vsw + (size_t)h * 1024 * 256);
      // scores per head (d=256, scale = 1/16)
      for (int h = 0; h < 2; ++h)
        gemm(qbf + h * 256, 512, kTsw + (size_t)h * 256 * 1024, nullptr,
             scores + (size_t)h * T * T, T, nullptr, 0, T, 256, T, 0.0625f, 0,
             kmask);
      k_softmax<<<2 * T, 256, 0, stream>>>(scores, probs, T);
      // o = probs @ v (per head) -> obf row-major
      for (int h = 0; h < 2; ++h)
        gemm(probs + (size_t)h * T * T, T, vsw + (size_t)h * 1024 * 256,
             nullptr, nullptr, 0, obf + h * 256, 512, T, T, 256, 1.0f, 0,
             nullptr);
      // out proj -> t2f
      k_wlin<<<ceil_div(nl, 256), 256, 0, stream>>>(F(pbase + 6), wbuf, 512, 512);
      gemm(obf, 512, wbuf, F(pbase + 7), t2f, 512, nullptr, 0,
           T, 512, 512, 1.0f, 0, nullptr);
    };

    // self-attention (key mask = sel)
    k_f2bf<<<ceil_div(TC, 256), 256, 0, stream>>>(yf, ybf, TC);
    mha(ybf, ybf, B0 + 0, selF);
    k_add_ln<<<T, 256, 0, stream>>>(yf, t2f, F(B0 + 16), F(B0 + 17));

    // cross-attention (memory = tok)
    k_f2bf<<<ceil_div(TC, 256), 256, 0, stream>>>(yf, ybf, TC);
    mha(ybf, tokbf, B0 + 8, nullptr);
    k_add_ln<<<T, 256, 0, stream>>>(yf, t2f, F(B0 + 18), F(B0 + 19));

    // FFN 512 -> 1024 -> 512
    k_f2bf<<<ceil_div(TC, 256), 256, 0, stream>>>(yf, ybf, TC);
    k_wlin<<<ceil_div(1024LL * 512, 256), 256, 0, stream>>>(F(B0 + 22), wbuf,
                                                            1024, 512);
    gemm(ybf, 512, wbuf, F(B0 + 23), nullptr, 0, ffnbf, 1024,
         T, 512, 1024, 1.0f, 1, nullptr);
    k_wlin<<<ceil_div(512LL * 1024, 256), 256, 0, stream>>>(F(B0 + 24), wbuf,
                                                            512, 1024);
    gemm(ffnbf, 1024, wbuf, F(B0 + 25), t2f, 512, nullptr, 0,
         T, 1024, 512, 1.0f, 0, nullptr);
    k_add_ln<<<T, 256, 0, stream>>>(yf, t2f, F(B0 + 20), F(B0 + 21));
  }

  // ---------------- scatter decoded tokens, reg head ----------------
  k_select<<<ceil_div(TC, 256), 256, 0, stream>>>(yf, tokf, selF, C, TC);
  k_f2bf<<<ceil_div(TC, 256), 256, 0, stream>>>(yf, ybf, TC);
  conv3(ybf, F(REG0 + 0), F(REG0 + 1), h1, 32, 32, 512, 256, 1);
  conv3(h1, F(REG0 + 2), F(REG0 + 3), h2, 32, 32, 256, 128, 1);
  k_conv1x1_c1<<<1024, 128, 0, stream>>>(h2, F(REG0 + 4), F(REG0 + 5),
                                         (float*)d_out, 128, 2, selF);
}
